// MultiHeadAttention_71631464563335
// MI455X (gfx1250) — matrix operand order repaired and checked by reading, not compiled, not run on hardware
//
#include <hip/hip_runtime.h>
#include <cstdint>

// Problem constants (from reference)
#define L_SEQ   1024
#define BATCH   8
#define DMODEL  1024
#define NHEAD   16
#define DHEAD   64
#define MROWS   (L_SEQ * BATCH)   // 8192 GEMM rows for (l,b) pairs

typedef __attribute__((ext_vector_type(16))) __bf16 v16bf;
typedef __attribute__((ext_vector_type(8)))  float  v8f;

union Frag16 {
    unsigned int w[8];
    v16bf v;
};

__device__ __forceinline__ unsigned short f32_to_bf16(float f) {
    unsigned int u = __float_as_uint(f);
    u += 0x7fffu + ((u >> 16) & 1u);   // round-to-nearest-even
    return (unsigned short)(u >> 16);
}

__device__ __forceinline__ unsigned int pack_bf16x2(float lo, float hi) {
    return (unsigned int)f32_to_bf16(lo) | ((unsigned int)f32_to_bf16(hi) << 16);
}

// A fragment (16x32 bf16): lane = M row; K-halves are 16B contiguous each.
__device__ __forceinline__ void load_fragA(Frag16& a, const unsigned short* row,
                                           int kk, int ka) {
    const uint4 lo = *(const uint4*)(row + kk + ka);
    const uint4 hi = *(const uint4*)(row + kk + ka + 16);
    a.w[0] = lo.x; a.w[1] = lo.y; a.w[2] = lo.z; a.w[3] = lo.w;
    a.w[4] = hi.x; a.w[5] = hi.y; a.w[6] = hi.z; a.w[7] = hi.w;
}

// B fragment (32x16 bf16): lane = N col; K-halves are 16B contiguous each.
__device__ __forceinline__ void load_fragB(Frag16& b, const unsigned short* row,
                                           int kk, int kb) {
    const uint4 b0 = *(const uint4*)(row + kk + (kb >> 1));
    const uint4 b1 = *(const uint4*)(row + kk + (kb >> 1) + 16);
    b.w[0] = b0.x; b.w[1] = b0.y; b.w[2] = b0.z; b.w[3] = b0.w;
    b.w[4] = b1.x; b.w[5] = b1.y; b.w[6] = b1.z; b.w[7] = b1.w;
}

// ---------------------------------------------------------------------------
// fp32 -> bf16 cast, 4 elements/thread (b128 load, b64 store)
// ---------------------------------------------------------------------------
__global__ void cast_f32_bf16_v4(const float4* __restrict__ in,
                                 uint2* __restrict__ out, int n4) {
    int i = blockIdx.x * blockDim.x + threadIdx.x;
    if (i < n4) {
        float4 v = in[i];
        uint2 o;
        o.x = pack_bf16x2(v.x, v.y);
        o.y = pack_bf16x2(v.z, v.w);
        out[i] = o;
    }
}

// ---------------------------------------------------------------------------
// Projection GEMM: A (MROWS x DMODEL bf16, rows = (l,b)) @ W^T (W: N x K bf16)
// vmode 0: store [b][h][l][dh]   (q, k)
// vmode 1: store [b][h][dh][l]   (v transposed, so PV B-operand is contiguous)
// One 16x16 output tile per wave; K-loop of 32 with v_wmma_f32_16x16x32_bf16.
// ---------------------------------------------------------------------------
__global__ void proj_gemm_kernel(const unsigned short* __restrict__ Ab,
                                 const unsigned short* __restrict__ Wb,
                                 unsigned short* __restrict__ Out,
                                 float scale, int vmode) {
    const int lane = threadIdx.x & 31;
    const int wid  = (blockIdx.x * blockDim.x + threadIdx.x) >> 5;
    const int NT_N = DMODEL / 16;          // 64 column tiles
    const int mt = wid / NT_N;
    const int nt = wid % NT_N;
    if (mt >= MROWS / 16) return;

    const int ln15 = lane & 15;
    const int ka = (lane >> 4) << 3;       // A frag K-split: 0 / 8
    const int kb = (lane >> 4) << 4;       // B frag K-split: 0 / 8 (used as kb >> 1)
    const unsigned short* __restrict__ arow = Ab + (size_t)(mt * 16 + ln15) * DMODEL;
    const unsigned short* __restrict__ brow = Wb + (size_t)(nt * 16 + ln15) * DMODEL;

    v8f acc = {};
    for (int kk = 0; kk < DMODEL; kk += 32) {
        Frag16 a, b;
        load_fragA(a, arow, kk, ka);
        load_fragB(b, brow, kk, kb);
        acc = __builtin_amdgcn_wmma_f32_16x16x32_bf16(
                  false, a.v, false, b.v, (short)0, acc, false, false);
    }

    const int n = nt * 16 + ln15;          // model dim e = h*64 + d
    const int h = n >> 6, d = n & 63;
#pragma unroll
    for (int r = 0; r < 8; ++r) {
        int m  = mt * 16 + r + ((lane >> 4) << 3);
        int l  = m >> 3;                    // row = l*BATCH + b
        int bb = m & 7;
        unsigned short val = f32_to_bf16(acc[r] * scale);
        if (vmode == 0)
            Out[(((size_t)(bb * NHEAD + h)) * L_SEQ + l) * DHEAD + d] = val;
        else
            Out[(((size_t)(bb * NHEAD + h)) * DHEAD + d) * L_SEQ + l] = val;
    }
}

// ---------------------------------------------------------------------------
// Attention: one block per (b, 16-row q-tile); loops over all 16 heads.
// LDS (dynamic, 162KB of the 320KB WGP pool):
//   sS   16x1024 fp32  : full score rows (exact one-pass softmax, no flash)
//   sCov 16x1024 fp32  : coverage accumulator across heads (no global atomics)
//   sP   16x1024 bf16  : normalized probs, A-operand of the PV WMMA
//   sQ   16x64   bf16  : Q tile, staged via global_load_async_to_lds_b64
// ---------------------------------------------------------------------------
#define SQ_LDS_OFF (16 * L_SEQ * 10)

__global__ void attn_kernel(const unsigned short* __restrict__ Qb,
                            const unsigned short* __restrict__ Kb,
                            const unsigned short* __restrict__ Vt,
                            const unsigned char*  __restrict__ Msk,
                            unsigned short* __restrict__ Ctx,
                            float* __restrict__ Cov) {
    extern __shared__ char smem[];
    float*          sS   = (float*)smem;                             // 64 KB
    float*          sCov = (float*)(smem + 16 * L_SEQ * 4);          // 64 KB
    unsigned short* sP   = (unsigned short*)(smem + 16 * L_SEQ * 8); // 32 KB
    unsigned short* sQ   = (unsigned short*)(smem + SQ_LDS_OFF);     // 2 KB

    const int tid = threadIdx.x, lane = tid & 31, wave = tid >> 5;
    const int qt = blockIdx.x & (L_SEQ / 16 - 1);
    const int b  = blockIdx.x / (L_SEQ / 16);
    const int q0 = qt * 16;
    const int ln15 = lane & 15;
    const int ka = (lane >> 4) << 3;
    const int kb = (lane >> 4) << 4;

    for (int i = tid; i < 16 * L_SEQ; i += 256) sCov[i] = 0.0f;
    __syncthreads();

    for (int h = 0; h < NHEAD; ++h) {
        const unsigned short* __restrict__ Qg =
            Qb + (((size_t)(b * NHEAD + h)) * L_SEQ + q0) * DHEAD;
        const unsigned short* __restrict__ Kg =
            Kb + ((size_t)(b * NHEAD + h)) * L_SEQ * DHEAD;
        const unsigned short* __restrict__ Vg =
            Vt + ((size_t)(b * NHEAD + h)) * DHEAD * L_SEQ;  // [dh][Lk]

        // ---- stage Q tile (2KB): one async b64 to LDS per lane ----
#if defined(__HIP_DEVICE_COMPILE__)
        {
            unsigned lds_off = __builtin_amdgcn_groupstaticsize()
                             + (unsigned)SQ_LDS_OFF + (unsigned)tid * 8u;
            const unsigned short* src = Qg + (size_t)tid * 4;
            asm volatile("global_load_async_to_lds_b64 %0, %1, off"
                         :: "v"(lds_off), "v"(src) : "memory");
            asm volatile("s_wait_asynccnt 0x0" ::: "memory");
        }
#else
        for (int i = tid; i < 16 * DHEAD / 2; i += 256)
            ((unsigned int*)sQ)[i] = ((const unsigned int*)Qg)[i];
#endif
        __syncthreads();

        // ---- S = Q K^T : each wave owns k-tiles wave, wave+8, ... ----
        for (int kt = wave; kt < L_SEQ / 16; kt += 8) {
            v8f acc = {};
            const unsigned short* arow = sQ + ln15 * DHEAD;
            const unsigned short* brow = Kg + (size_t)(kt * 16 + ln15) * DHEAD;
#pragma unroll
            for (int kk = 0; kk < DHEAD; kk += 32) {
                Frag16 a, bf;
                load_fragA(a, arow, kk, ka);   // ds_load_b128 x2
                load_fragB(bf, brow, kk, kb);  // global_load_b128 x2
                acc = __builtin_amdgcn_wmma_f32_16x16x32_bf16(
                          false, a.v, false, bf.v, (short)0, acc, false, false);
            }
#pragma unroll
            for (int r = 0; r < 8; ++r) {
                int m = r + ((lane >> 4) << 3);
                sS[m * L_SEQ + kt * 16 + ln15] = acc[r];
            }
        }
        __syncthreads();

        // ---- softmax + coverage: wave handles rows 2w, 2w+1 ----
        for (int rr = 0; rr < 2; ++rr) {
            int m = wave * 2 + rr;
            const unsigned char* mk = Msk + ((size_t)b * L_SEQ + (q0 + m)) * L_SEQ;
            float mx = -3.0e38f;
            for (int j = lane; j < L_SEQ; j += 32) {
                float s = sS[m * L_SEQ + j];
                if (mk[j]) s = -INFINITY;
                sS[m * L_SEQ + j] = s;
                mx = fmaxf(mx, s);
            }
#pragma unroll
            for (int off = 16; off >= 1; off >>= 1)
                mx = fmaxf(mx, __shfl_xor(mx, off, 32));
            float sum = 0.0f;
            for (int j = lane; j < L_SEQ; j += 32) {
                float e = __expf(sS[m * L_SEQ + j] - mx);
                sS[m * L_SEQ + j] = e;
                sum += e;
            }
#pragma unroll
            for (int off = 16; off >= 1; off >>= 1)
                sum += __shfl_xor(sum, off, 32);
            float inv = 1.0f / sum;
            for (int j = lane; j < L_SEQ; j += 32) {
                float p = sS[m * L_SEQ + j] * inv;
                sP[m * L_SEQ + j] = f32_to_bf16(p);
                sCov[m * L_SEQ + j] += p * (1.0f / NHEAD);
            }
        }
        __syncthreads();

        // ---- O = P V : waves 0..3 each own one 16-wide slice of dh ----
        if (wave < 4) {
            const int n0 = wave * 16;
            v8f acc = {};
            const unsigned short* arow = sP + ln15 * L_SEQ;
            const unsigned short* brow = Vg + (size_t)(n0 + ln15) * L_SEQ; // V^T row
            for (int kk = 0; kk < L_SEQ; kk += 32) {
                Frag16 a, bf;
                load_fragA(a, arow, kk, ka);   // ds_load_b128 x2
                load_fragB(bf, brow, kk, kb);  // global_load_b128 x2 (contiguous!)
                acc = __builtin_amdgcn_wmma_f32_16x16x32_bf16(
                          false, a.v, false, bf.v, (short)0, acc, false, false);
            }
#pragma unroll
            for (int r = 0; r < 8; ++r) {
                int m = r + ((lane >> 4) << 3);
                int l = q0 + m;
                Ctx[((size_t)l * BATCH + b) * DMODEL + h * DHEAD + n0 + ln15] =
                    f32_to_bf16(acc[r]);
            }
        }
        __syncthreads();
    }

    // coverage out: (B, Lq, Lk) fp32
    float* cv = Cov + ((size_t)b * L_SEQ + q0) * L_SEQ;
    for (int i = tid; i < 16 * L_SEQ; i += 256) cv[i] = sCov[i];
}

// ---------------------------------------------------------------------------
// Output GEMM: ctx (MROWS x DMODEL bf16) @ Wo^T -> d_out fp32 (same row order)
// ---------------------------------------------------------------------------
__global__ void out_gemm_kernel(const unsigned short* __restrict__ Ab,
                                const unsigned short* __restrict__ Wb,
                                float* __restrict__ Out) {
    const int lane = threadIdx.x & 31;
    const int wid  = (blockIdx.x * blockDim.x + threadIdx.x) >> 5;
    const int NT_N = DMODEL / 16;
    const int mt = wid / NT_N;
    const int nt = wid % NT_N;
    if (mt >= MROWS / 16) return;

    const int ln15 = lane & 15;
    const int ka = (lane >> 4) << 3;
    const int kb = (lane >> 4) << 4;
    const unsigned short* __restrict__ arow = Ab + (size_t)(mt * 16 + ln15) * DMODEL;
    const unsigned short* __restrict__ brow = Wb + (size_t)(nt * 16 + ln15) * DMODEL;

    v8f acc = {};
    for (int kk = 0; kk < DMODEL; kk += 32) {
        Frag16 a, b;
        load_fragA(a, arow, kk, ka);
        load_fragB(b, brow, kk, kb);
        acc = __builtin_amdgcn_wmma_f32_16x16x32_bf16(
                  false, a.v, false, b.v, (short)0, acc, false, false);
    }

    int n = nt * 16 + ln15;
#pragma unroll
    for (int r = 0; r < 8; ++r) {
        int m = mt * 16 + r + ((lane >> 4) << 3);
        Out[(size_t)m * DMODEL + n] = acc[r];
    }
}

// ---------------------------------------------------------------------------
extern "C" void kernel_launch(void* const* d_in, const int* in_sizes, int n_in,
                              void* d_out, int out_size, void* d_ws, size_t ws_size,
                              hipStream_t stream) {
    (void)in_sizes; (void)n_in; (void)out_size; (void)ws_size;

    const float*         query = (const float*)d_in[0];
    const float*         key_  = (const float*)d_in[1];
    const float*         value = (const float*)d_in[2];
    const unsigned char* mask  = (const unsigned char*)d_in[3];
    const float*         Wq    = (const float*)d_in[4];
    const float*         Wk    = (const float*)d_in[5];
    const float*         Wv    = (const float*)d_in[6];
    const float*         Wo    = (const float*)d_in[7];

    // workspace layout (bf16 elements):
    //   weights 4x1M | x(q,k,v) 3x8M | proj q/k/v 3x8M | ctx 8M  = 60M elems (120MB)
    const size_t WE = (size_t)DMODEL * DMODEL;                 // 1M
    const size_t XE = (size_t)MROWS * DMODEL;                  // 8M
    unsigned short* Wqb = (unsigned short*)d_ws;
    unsigned short* Wkb = Wqb + WE;
    unsigned short* Wvb = Wkb + WE;
    unsigned short* Wob = Wvb + WE;
    unsigned short* xq  = Wob + WE;
    unsigned short* xk  = xq + XE;
    unsigned short* xv  = xk + XE;
    unsigned short* qb  = xv + XE;
    unsigned short* kb  = qb + XE;
    unsigned short* vt  = kb + XE;   // [b][h][dh][l]
    unsigned short* ctx = vt + XE;

    float* outp = (float*)d_out;
    float* covp = outp + (size_t)L_SEQ * BATCH * DMODEL;

    const int WB4 = (int)(WE / 4), XB4 = (int)(XE / 4);
    cast_f32_bf16_v4<<<WB4 / 256, 256, 0, stream>>>((const float4*)Wq, (uint2*)Wqb, WB4);
    cast_f32_bf16_v4<<<WB4 / 256, 256, 0, stream>>>((const float4*)Wk, (uint2*)Wkb, WB4);
    cast_f32_bf16_v4<<<WB4 / 256, 256, 0, stream>>>((const float4*)Wv, (uint2*)Wvb, WB4);
    cast_f32_bf16_v4<<<WB4 / 256, 256, 0, stream>>>((const float4*)Wo, (uint2*)Wob, WB4);
    cast_f32_bf16_v4<<<XB4 / 256, 256, 0, stream>>>((const float4*)query, (uint2*)xq, XB4);
    cast_f32_bf16_v4<<<XB4 / 256, 256, 0, stream>>>((const float4*)key_,  (uint2*)xk, XB4);
    cast_f32_bf16_v4<<<XB4 / 256, 256, 0, stream>>>((const float4*)value, (uint2*)xv, XB4);

    const int GB = (MROWS / 16) * (DMODEL / 16) / 8;  // 8 tile-waves per block
    proj_gemm_kernel<<<GB, 256, 0, stream>>>(xq, Wqb, qb, 0.125f, 0); // 1/sqrt(64)
    proj_gemm_kernel<<<GB, 256, 0, stream>>>(xk, Wkb, kb, 1.0f, 0);
    proj_gemm_kernel<<<GB, 256, 0, stream>>>(xv, Wvb, vt, 1.0f, 1);  // transposed

    const int SMEM = SQ_LDS_OFF + 16 * DHEAD * 2;     // 165888 B (< 320KB WGP LDS)
    hipFuncSetAttribute((const void*)attn_kernel,
                        hipFuncAttributeMaxDynamicSharedMemorySize, SMEM);
    attn_kernel<<<BATCH * (L_SEQ / 16), 256, SMEM, stream>>>(qb, kb, vt, mask, ctx, covp);

    out_gemm_kernel<<<GB, 256, 0, stream>>>(ctx, Wob, outp);
}
